// TAM_71390946394310
// MI455X (gfx1250) — matrix operand order repaired and checked by reading, not compiled, not run on hardware
//
#include <hip/hip_runtime.h>
#include <hip/hip_bf16.h>
#include <math.h>

// ---------------- problem constants (match reference setup_inputs) ----------
constexpr int N_  = 64;
constexpr int T_  = 300;
constexpr int C_  = 64;
constexpr int V_  = 25;
constexpr int K_  = 3;
constexpr int NC_ = N_ * C_;            // 4096 rows for the G-branch GEMM
constexpr int H_  = 2 * T_;             // 600 hidden dim
constexpr int KT  = 10;                 // K-tiles of 32 (300 padded to 320)
constexpr int JT  = 38;                 // column tiles of 16 (600 padded to 608)
constexpr int HCOL = 609;               // H-tile LDS stride (odd -> conflict free)
constexpr float BN_SCALE_C = 0.99999500003749978f; // 1/sqrt(1+1e-5)

typedef __attribute__((ext_vector_type(16))) __bf16 v16bf;
typedef __attribute__((ext_vector_type(8)))  float  v8f;

__device__ __forceinline__ __bf16 f2bf(float f) {
    union { unsigned u; float f; } uf; uf.f = f;
    unsigned u = uf.u + 0x7FFFu + ((uf.u >> 16) & 1u);   // round-to-nearest-even
    union { unsigned short s; __bf16 b; } ub;
    ub.s = (unsigned short)(u >> 16);
    return ub.b;
}

// ---------------------------------------------------------------------------
// Kernel 1: pooled[(n*C+c)*T + t] = mean_v x[((n*T+t)*C+c)*V + v]
// Coalesced: stage 256 consecutive (n,t,c) rows (256*25 floats) in LDS.
// ---------------------------------------------------------------------------
__global__ __launch_bounds__(256)
void tam_pool_kernel(const float* __restrict__ x, float* __restrict__ pooled) {
    __shared__ float buf[256 * V_];
    const int tid  = threadIdx.x;
    const int base = blockIdx.x * 256;            // first (n,t,c) group id
    const long gbase = (long)base * V_;
    for (int i = tid; i < 256 * V_; i += 256)
        buf[i] = x[gbase + i];
    __syncthreads();
    float s = 0.f;
    #pragma unroll
    for (int v = 0; v < V_; ++v) s += buf[tid * V_ + v];
    s *= (1.0f / V_);
    const int g = base + tid;                     // g = (n*T + t)*C + c
    const int c = g % C_;
    const int t = (g / C_) % T_;
    const int n = g / (C_ * T_);
    pooled[(n * C_ + c) * T_ + t] = s;
}

// ---------------------------------------------------------------------------
// Kernel 1b: pre-swizzle g_w1 (f32 [600][300]) into bf16 WMMA B-fragment
// order: Bsw[((jt*KT + kk)*32 + lane)*16 + e]
//   lane n = L&15 (column j = jt*16+n), hi = L>>4, element e -> K = kk*32 + ((e<8)?0:16) + hi*8 + (e&7)
// Zero-padded for j >= 600 and K >= 300. One-time 194,560-element conversion.
// ---------------------------------------------------------------------------
__global__ __launch_bounds__(256)
void tam_bprep_kernel(const float* __restrict__ gw1, __bf16* __restrict__ Bsw) {
    const int idx = blockIdx.x * 256 + threadIdx.x;       // [0, JT*KT*512)
    const int e  = idx & 15;
    const int L  = (idx >> 4) & 31;
    const int fk = idx >> 9;                              // jt*KT + kk
    const int jt = fk / KT, kk = fk % KT;
    const int n  = L & 15, hi = L >> 4;
    const int j  = jt * 16 + n;
    const int k  = kk * 32 + ((e < 8) ? 0 : 16) + hi * 8 + (e & 7);
    const float v = (j < H_ && k < T_) ? gw1[j * T_ + k] : 0.f;
    Bsw[idx] = f2bf(v);
}

// ---------------------------------------------------------------------------
// Kernel 2: G branch.
//   h = relu(BN_SCALE * pooled[4096,300] @ g_w1[600,300]^T)   (WMMA bf16)
//   conv_kernel = softmax(h @ g_w2[3,600]^T)                  (VALU)
// One wave per 16-row tile. A fragments hoisted to registers (reused across
// all 38 column tiles); inner loop = one 32B global B load + one WMMA.
// ---------------------------------------------------------------------------
__global__ __launch_bounds__(32)
void tam_gbranch_kernel(const float* __restrict__ pooled,
                        const v16bf* __restrict__ Bsw,
                        const float* __restrict__ gw2,
                        float* __restrict__ ck) {
    __shared__ __attribute__((aligned(32))) __bf16 Aswz[KT * 32 * 16]; // 10,240 B
    __shared__ float Htile[16][HCOL];                                  // 38,976 B

    const int lane    = threadIdx.x;
    const int rowBase = blockIdx.x * 16;          // global row in [0, 4096)

    // Stage A (16 rows of pooled) into LDS directly in WMMA fragment order:
    // A 16x32 bf16 layout: lane m holds row m; VGPR i (elements 2i,2i+1) holds
    // K = ((i<4)?0:16) + hi*8 + 2*(i&3) (+1).
    for (int idx = lane; idx < KT * 32 * 16; idx += 32) {
        const int kk = idx >> 9;
        const int L  = (idx >> 4) & 31;
        const int e  = idx & 15;
        const int i  = e >> 1, b = e & 1;
        const int m  = L & 15, hi = L >> 4;
        const int k  = kk * 32 + ((i < 4) ? 0 : 16) + hi * 8 + 2 * (i & 3) + b;
        const float v = (k < T_) ? pooled[(rowBase + m) * T_ + k] : 0.f;
        Aswz[idx] = f2bf(v);
    }
    __syncthreads();

    // Hoist all 10 A fragments into registers (10 x 8 VGPRs).
    v16bf afrag[KT];
    const v16bf* As = (const v16bf*)Aswz;
    #pragma unroll
    for (int kk = 0; kk < KT; ++kk) afrag[kk] = As[kk * 32 + lane];

    const int m  = lane & 15;                     // B/D column within tile
    const int hi = lane >> 4;

    for (int jt = 0; jt < JT; ++jt) {
        v8f acc = {0.f, 0.f, 0.f, 0.f, 0.f, 0.f, 0.f, 0.f};
        const v16bf* Bt = Bsw + (size_t)jt * KT * 32 + lane;
        #pragma unroll
        for (int kk = 0; kk < KT; ++kk) {
            const v16bf b = Bt[kk * 32];          // one aligned 32B load
            acc = __builtin_amdgcn_wmma_f32_16x16x32_bf16(
                false, afrag[kk], false, b, (short)0, acc, false, false);
        }
        // D layout: lane -> column m; VGPR r -> row r + hi*8
        const int col = jt * 16 + m;
        #pragma unroll
        for (int r = 0; r < 8; ++r) {
            if (col < H_) Htile[r + hi * 8][col] = fmaxf(acc[r] * BN_SCALE_C, 0.f);
        }
    }
    __syncthreads();

    // Second linear (600 -> 3) + softmax; lanes 0..15 each own one row.
    if (lane < 16) {
        float a0 = 0.f, a1 = 0.f, a2 = 0.f;
        for (int jj = 0; jj < H_; ++jj) {
            const float hv = Htile[lane][jj];
            a0 += hv * gw2[jj];
            a1 += hv * gw2[H_ + jj];
            a2 += hv * gw2[2 * H_ + jj];
        }
        const float mx  = fmaxf(a0, fmaxf(a1, a2));
        const float e0  = __expf(a0 - mx);
        const float e1  = __expf(a1 - mx);
        const float e2  = __expf(a2 - mx);
        const float inv = 1.f / (e0 + e1 + e2);
        float* o = ck + (rowBase + lane) * 3;
        o[0] = e0 * inv; o[1] = e1 * inv; o[2] = e2 * inv;
    }
}

// ---------------------------------------------------------------------------
// Kernel 3: L branch.
//   z = relu(BN_SCALE * conv1d(pooled, l_w1, pad=1))   [N,16,T]
//   local_act = sigmoid(einsum("nct,oc->not", z, l_w2)) [N,C,T]
// Block = (n, 75-timestep tile); pooled window + weights staged in LDS.
// ---------------------------------------------------------------------------
constexpr int TT_ = 75;   // T tile (300/4)

__global__ __launch_bounds__(256)
void tam_lbranch_kernel(const float* __restrict__ pooled,
                        const float* __restrict__ lw1,
                        const float* __restrict__ lw2,
                        float* __restrict__ la) {
    __shared__ float pw[C_][TT_ + 2];        // 19,712 B
    __shared__ float zs[16][TT_ + 2];        //  4,928 B
    __shared__ float w1s[16 * C_ * K_];      // 12,288 B
    __shared__ float w2s[C_ * 16];           //  4,096 B

    const int tid = threadIdx.x;
    const int n   = blockIdx.y;
    const int t0  = blockIdx.x * TT_;

    for (int i = tid; i < C_ * (TT_ + 2); i += 256) {
        const int c  = i / (TT_ + 2);
        const int tt = i % (TT_ + 2);
        const int t  = t0 - 1 + tt;
        pw[c][tt] = (t >= 0 && t < T_) ? pooled[(n * C_ + c) * T_ + t] : 0.f;
    }
    for (int i = tid; i < 16 * C_ * K_; i += 256) w1s[i] = lw1[i];
    for (int i = tid; i < C_ * 16;      i += 256) w2s[i] = lw2[i];
    __syncthreads();

    // Phase A: z[o][tt]
    for (int item = tid; item < 16 * TT_; item += 256) {
        const int o  = item / TT_;
        const int tt = item % TT_;
        float acc = 0.f;
        #pragma unroll 4
        for (int c = 0; c < C_; ++c) {
            const float* w = &w1s[(o * C_ + c) * K_];
            acc += pw[c][tt]     * w[0];
            acc += pw[c][tt + 1] * w[1];
            acc += pw[c][tt + 2] * w[2];
        }
        zs[o][tt] = fmaxf(acc * BN_SCALE_C, 0.f);
    }
    __syncthreads();

    // Phase B: local_act[n][oc][t0+tt]
    for (int item = tid; item < C_ * TT_; item += 256) {
        const int oc = item / TT_;
        const int tt = item % TT_;
        float acc = 0.f;
        #pragma unroll
        for (int o = 0; o < 16; ++o) acc += zs[o][tt] * w2s[oc * 16 + o];
        la[(n * C_ + oc) * T_ + (t0 + tt)] = 1.f / (1.f + __expf(-acc));
    }
}

// ---------------------------------------------------------------------------
// Kernel 4: gate + per-(n,c) depthwise temporal conv (the bandwidth kernel).
//   out[n,t,c,v] = sum_k x[n,t+k-1,c,v] * la[n,c,t+k-1] * ck[n,c,k]
// One thread per element; x/out streamed coalesced; ±1 timestep taps hit L2.
// ---------------------------------------------------------------------------
__global__ __launch_bounds__(256)
void tam_final_kernel(const float* __restrict__ x,
                      const float* __restrict__ la,
                      const float* __restrict__ ck,
                      float* __restrict__ out) {
    const long idx = (long)blockIdx.x * 256 + threadIdx.x;
    const int c = (int)((idx / V_) % C_);
    const int t = (int)((idx / (V_ * C_)) % T_);
    const int n = (int)(idx / ((long)V_ * C_ * T_));

    const float* ckp = ck + (n * C_ + c) * K_;
    const float* lap = la + (n * C_ + c) * T_;
    const float* xp  = x + idx;
    constexpr int TS = C_ * V_;                   // stride between timesteps

    float acc = xp[0] * lap[t] * ckp[1];
    if (t > 0)      acc += xp[-TS] * lap[t - 1] * ckp[0];
    if (t < T_ - 1) acc += xp[ TS] * lap[t + 1] * ckp[2];
    out[idx] = acc;
}

// ---------------------------------------------------------------------------
extern "C" void kernel_launch(void* const* d_in, const int* in_sizes, int n_in,
                              void* d_out, int out_size, void* d_ws, size_t ws_size,
                              hipStream_t stream) {
    const float* x   = (const float*)d_in[0];   // [N,T,C,V]
    const float* gw1 = (const float*)d_in[1];   // [600,300]
    const float* gw2 = (const float*)d_in[2];   // [3,600]
    const float* lw1 = (const float*)d_in[3];   // [16,64,3]
    const float* lw2 = (const float*)d_in[4];   // [64,16]
    float* out = (float*)d_out;                 // [N,T,C,V]

    // Workspace layout (~10.3 MB)
    float* pooled = (float*)d_ws;                       // [N*C, T]
    float* la     = pooled + (size_t)NC_ * T_;          // [N*C, T]
    float* ck     = la + (size_t)NC_ * T_;              // [N*C, 3]
    __bf16* Bsw   = (__bf16*)(ck + (size_t)NC_ * K_);   // [JT*KT*32*16] bf16
                                                        // (offset 32B-aligned)

    // 1) pooling over V
    tam_pool_kernel<<<(N_ * T_ * C_) / 256, 256, 0, stream>>>(x, pooled);

    // 1b) one-time g_w1 -> bf16 fragment swizzle (JT*KT*512 = 194,560 elems)
    tam_bprep_kernel<<<(JT * KT * 512) / 256, 256, 0, stream>>>(gw1, Bsw);

    // 2) G branch (WMMA GEMM + softmax) -> per-(n,c) kernels
    tam_gbranch_kernel<<<NC_ / 16, 32, 0, stream>>>(pooled, (const v16bf*)Bsw,
                                                    gw2, ck);

    // 3) L branch -> local activation gate
    tam_lbranch_kernel<<<dim3(T_ / TT_, N_), 256, 0, stream>>>(pooled, lw1, lw2, la);

    // 4) gate + depthwise temporal conv, write output
    const long total = (long)N_ * T_ * C_ * V_;
    tam_final_kernel<<<(unsigned)(total / 256), 256, 0, stream>>>(x, la, ck, out);
}
